// GATModel_83468394430530
// MI455X (gfx1250) — compile-verified, matrix operand order checked by reading, not run on hardware
//
#include <hip/hip_runtime.h>
#include <math.h>

typedef __attribute__((ext_vector_type(16))) _Float16 v16h;
typedef __attribute__((ext_vector_type(8)))  float    v8f;

#define N_NODES 50000
#define N_EDGES 800000
#define DIM 128
#define E_TOT (N_EDGES + N_NODES)   // original edges + self loops
#define NEG_SLOPE 0.2f
#define ROWS_PER_BLOCK 64           // 4 row-tiles per wave

// B-fragment chunk layout in LDS:
//   chunk q*128+col, q = (k0/32)*2 + khalf, holds the 16 halves
//   W[kb..kb+7][col], W[kb+16..kb+23][col]  (kb = k0 + khalf*8)
// Chunk stride padded to 24 halves (48 B) -> conflict-free ds_load_b128.
#define CHUNK_STRIDE 24
#define N_CHUNKS     (8 * DIM)      // 4 k-steps * 2 k-halves * 128 cols

// ---------------------------------------------------------------------------
// GEMM: H[N,128] = A[N,128] @ W[128,128], f16 inputs, f32 accum via WMMA.
// Block = 256 threads = 8 waves; block owns 64 rows (4 row-tiles / wave),
// wave w owns the 16-col tile at 16w. B fragments hoisted to registers and
// reused across the 4 row tiles -> 16 x v_wmma_f32_16x16x32_f16 per wave,
// W->LDS staging amortized 4x.
// ---------------------------------------------------------------------------
__global__ __launch_bounds__(256) void gat_gemm_wmma(
    const float* __restrict__ A, const float* __restrict__ W,
    float* __restrict__ H, int n_rows)
{
    __shared__ _Float16 Wl[N_CHUNKS * CHUNK_STRIDE];   // 48 KB

    // ---- Stage W -> LDS, converted to f16 and pre-swizzled into fragments.
    // Thread t handles chunks t, t+256, t+512, t+768. Global reads coalesce
    // across the 128-column group; writes are two packed ds_store_b128.
    const int t = threadIdx.x;
    #pragma unroll
    for (int i = 0; i < N_CHUNKS / 256; ++i) {
        const int c   = t + i * 256;
        const int col = c & (DIM - 1);
        const int q   = c >> 7;              // 0..7
        const int kb  = (q >> 1) * 32 + (q & 1) * 8;
        v16h frag;
        #pragma unroll
        for (int e = 0; e < 8; ++e) frag[e]     = (_Float16)W[(kb + e) * DIM + col];
        #pragma unroll
        for (int e = 0; e < 8; ++e) frag[8 + e] = (_Float16)W[(kb + 16 + e) * DIM + col];
        *(__attribute__((address_space(3))) v16h*)(&Wl[c * CHUNK_STRIDE]) = frag;
    }
    __syncthreads();

    const int wave = t >> 5;
    const int lane = t & 31;
    const int lm   = lane & 15;     // row (A) / col (B) within tile
    const int lh   = lane >> 4;     // K-half selector
    const int j0   = wave * 16;     // col tile base

    // Hoist the wave's four B fragments (one per K-step) into registers.
    v16h bfrag[4];
    #pragma unroll
    for (int k = 0; k < 4; ++k) {
        const int ci = (k * 2 + lh) * DIM + (j0 + lm);
        bfrag[k] = *(__attribute__((address_space(3))) const v16h*)
                        (&Wl[ci * CHUNK_STRIDE]);
    }

    #pragma unroll
    for (int rt = 0; rt < ROWS_PER_BLOCK / 16; ++rt) {
        const int i0 = blockIdx.x * ROWS_PER_BLOCK + rt * 16;  // row tile base
        if (i0 >= n_rows) break;        // uniform (block-level) guard

        v8f c = {};
        #pragma unroll
        for (int k = 0; k < 4; ++k) {
            const int kb = k * 32 + lh * 8;
            // A fragment: lane (lm,lh) -> row i0+lm, K = kb..kb+7, kb+16..kb+23
            v16h a;
            const float* ap = A + (i0 + lm) * DIM + kb;
            #pragma unroll
            for (int e = 0; e < 8; ++e) a[e]     = (_Float16)ap[e];
            #pragma unroll
            for (int e = 0; e < 8; ++e) a[8 + e] = (_Float16)ap[16 + e];

            c = __builtin_amdgcn_wmma_f32_16x16x32_f16(
                    false, a, false, bfrag[k], (short)0, c, false, false);
        }

        // C layout: lanes 0-15 VGPR r -> (M=r,N=lane); lanes 16-31 -> (M=8+r)
        float* hp = H + (i0 + lh * 8) * DIM + (j0 + lm);
        #pragma unroll
        for (int r = 0; r < 8; ++r) hp[r * DIM] = c[r];
    }
}

// ---------------------------------------------------------------------------
// Per-node attention logits: al_src[n] = h[n].a_src, al_dst[n] = h[n].a_dst
// One wave per node, lane handles one float4 of the 128-dim row.
// ---------------------------------------------------------------------------
__global__ __launch_bounds__(256) void gat_logits(
    const float* __restrict__ h, const float* __restrict__ a_src,
    const float* __restrict__ a_dst, float* __restrict__ al_src,
    float* __restrict__ al_dst)
{
    const int node = (blockIdx.x * blockDim.x + threadIdx.x) >> 5;
    const int lane = threadIdx.x & 31;
    if (node >= N_NODES) return;

    const float4 hv = ((const float4*)(h + node * DIM))[lane];
    const float4 sv = ((const float4*)a_src)[lane];
    const float4 dv = ((const float4*)a_dst)[lane];
    float ds = hv.x * sv.x + hv.y * sv.y + hv.z * sv.z + hv.w * sv.w;
    float dd = hv.x * dv.x + hv.y * dv.y + hv.z * dv.z + hv.w * dv.w;
    #pragma unroll
    for (int off = 16; off; off >>= 1) {
        ds += __shfl_xor(ds, off, 32);
        dd += __shfl_xor(dd, off, 32);
    }
    if (lane == 0) { al_src[node] = ds; al_dst[node] = dd; }
}

// ---------------------------------------------------------------------------
// Init per-layer state: agg=0 (N*128), m=-big, denom=0 (first N threads).
// ---------------------------------------------------------------------------
__global__ __launch_bounds__(256) void gat_init(
    float* __restrict__ agg, float* __restrict__ m, float* __restrict__ den)
{
    const int i = blockIdx.x * blockDim.x + threadIdx.x;
    if (i < N_NODES * DIM) agg[i] = 0.0f;
    if (i < N_NODES) { m[i] = -3.0e38f; den[i] = 0.0f; }
}

__device__ __forceinline__ void edge_ids(int e, const int* src, const int* dst,
                                         int& s, int& d)
{
    if (e < N_EDGES) { s = src[e]; d = dst[e]; }
    else             { s = e - N_EDGES; d = s; }      // self loop
}

// ---------------------------------------------------------------------------
// Edge pass 1: raw logit + LeakyReLU, segment max into m[dst]
// (native float atomicMax -> global_atomic_max_num_f32).
// ---------------------------------------------------------------------------
__global__ __launch_bounds__(256) void gat_edge_max(
    const int* __restrict__ src, const int* __restrict__ dst,
    const float* __restrict__ al_src, const float* __restrict__ al_dst,
    float* __restrict__ ev, float* __restrict__ m)
{
    const int e = blockIdx.x * blockDim.x + threadIdx.x;
    if (e >= E_TOT) return;
    int s, d; edge_ids(e, src, dst, s, d);
    float v = al_src[s] + al_dst[d];
    v = (v >= 0.0f) ? v : NEG_SLOPE * v;
    ev[e] = v;
    atomicMax(&m[d], v);
}

// ---------------------------------------------------------------------------
// Edge pass 2: ex = exp(e - m[dst]) (in place), segment sum into denom[dst].
// ---------------------------------------------------------------------------
__global__ __launch_bounds__(256) void gat_edge_exp(
    const int* __restrict__ src, const int* __restrict__ dst,
    float* __restrict__ ev, const float* __restrict__ m,
    float* __restrict__ den)
{
    const int e = blockIdx.x * blockDim.x + threadIdx.x;
    if (e >= E_TOT) return;
    int s, d; edge_ids(e, src, dst, s, d);
    const float v = __expf(ev[e] - m[d]);
    ev[e] = v;
    atomicAdd(&den[d], v);
}

// ---------------------------------------------------------------------------
// Edge pass 3: agg[dst] += (ex/denom[dst]) * h[src]. One wave per edge,
// lane handles one float4 (coalesced 512B row gather, 4 f32 atomics/lane).
// ---------------------------------------------------------------------------
__global__ __launch_bounds__(256) void gat_scatter(
    const int* __restrict__ src, const int* __restrict__ dst,
    const float* __restrict__ ev, const float* __restrict__ den,
    const float* __restrict__ h, float* __restrict__ agg)
{
    const int e    = (blockIdx.x * blockDim.x + threadIdx.x) >> 5;
    const int lane = threadIdx.x & 31;
    if (e >= E_TOT) return;
    int s, d; edge_ids(e, src, dst, s, d);
    const float alpha = ev[e] / den[d];
    const float4 hv = ((const float4*)(h + s * DIM))[lane];
    float* ap = agg + d * DIM + lane * 4;
    atomicAdd(ap + 0, alpha * hv.x);
    atomicAdd(ap + 1, alpha * hv.y);
    atomicAdd(ap + 2, alpha * hv.z);
    atomicAdd(ap + 3, alpha * hv.w);
}

// ---------------------------------------------------------------------------
// Finalize: out = relu(agg + b). May run in place (out == agg).
// ---------------------------------------------------------------------------
__global__ __launch_bounds__(256) void gat_final(
    const float* __restrict__ agg, const float* __restrict__ b,
    float* __restrict__ out)
{
    const int i = blockIdx.x * blockDim.x + threadIdx.x;
    if (i >= N_NODES * DIM) return;
    out[i] = fmaxf(agg[i] + b[i & (DIM - 1)], 0.0f);
}

// ---------------------------------------------------------------------------
extern "C" void kernel_launch(void* const* d_in, const int* in_sizes, int n_in,
                              void* d_out, int out_size, void* d_ws, size_t ws_size,
                              hipStream_t stream)
{
    const float* x   = (const float*)d_in[0];
    const int*   ei  = (const int*)  d_in[1];   // [2, E] flat, int32
    // d_in[2] = edge_weight: ignored (GATConv edge_dim=None)
    const float* W1  = (const float*)d_in[3];
    const float* as1 = (const float*)d_in[4];
    const float* ad1 = (const float*)d_in[5];
    const float* b1  = (const float*)d_in[6];
    const float* W2  = (const float*)d_in[7];
    const float* as2 = (const float*)d_in[8];
    const float* ad2 = (const float*)d_in[9];
    const float* b2  = (const float*)d_in[10];
    float* out = (float*)d_out;

    const int* src = ei;
    const int* dst = ei + N_EDGES;

    // Workspace layout (floats): ~56 MB total
    float* ws  = (float*)d_ws;
    float* h   = ws;                           // N*128
    float* act = h   + (size_t)N_NODES * DIM;  // N*128  (layer-1 output)
    float* als = act + (size_t)N_NODES * DIM;  // N
    float* ald = als + N_NODES;                // N
    float* m   = ald + N_NODES;                // N
    float* den = m   + N_NODES;                // N
    float* ev  = den + N_NODES;                // E_TOT

    const int gridGemm  = (N_NODES + ROWS_PER_BLOCK - 1) / ROWS_PER_BLOCK; // 782
    const int gridNode  = (N_NODES * 32 + 255) / 256;      // wave per node
    const int gridElem  = (N_NODES * DIM + 255) / 256;
    const int gridEdge  = (E_TOT + 255) / 256;
    const int gridEdgeW = (E_TOT * 32 + 255) / 256;        // wave per edge

    auto layer = [&](const float* in_act, const float* W, const float* a_s,
                     const float* a_d, const float* b, float* agg_out) {
        gat_gemm_wmma<<<gridGemm, 256, 0, stream>>>(in_act, W, h, N_NODES);
        gat_logits  <<<gridNode, 256, 0, stream>>>(h, a_s, a_d, als, ald);
        gat_init    <<<gridElem, 256, 0, stream>>>(agg_out, m, den);
        gat_edge_max<<<gridEdge, 256, 0, stream>>>(src, dst, als, ald, ev, m);
        gat_edge_exp<<<gridEdge, 256, 0, stream>>>(src, dst, ev, m, den);
        gat_scatter <<<gridEdgeW, 256, 0, stream>>>(src, dst, ev, den, h, agg_out);
        gat_final   <<<gridElem, 256, 0, stream>>>(agg_out, b, agg_out);
    };

    layer(x,   W1, as1, ad1, b1, act);   // layer 1 -> act (in ws)
    layer(act, W2, as2, ad2, b2, out);   // layer 2 -> d_out
}
